// RarityNetwork_55508157333685
// MI455X (gfx1250) — matrix operand order repaired and checked by reading, not compiled, not run on hardware
//
#include <hip/hip_runtime.h>
#include <hip/hip_bf16.h>

#define EPSV   1e-8f
#define NPIX   3136      // 56*56
#define NBINS  11
#define NMAPS  80        // 10 layers * 8 batch
#define RPIX   57600     // 240*240

typedef __attribute__((ext_vector_type(16))) _Float16 v16h;
typedef __attribute__((ext_vector_type(8)))  float    v8f;
typedef __attribute__((ext_vector_type(4)))  int      v4i;

// gfx1250 async global->LDS path (ASYNCcnt), guarded so either toolchain compiles.
#if defined(__has_builtin)
#if __has_builtin(__builtin_amdgcn_global_load_async_to_lds_b128) && \
    __has_builtin(__builtin_amdgcn_s_wait_asynccnt)
#define USE_ASYNC_LDS 1
#endif
#endif
#ifndef USE_ASYNC_LDS
#define USE_ASYNC_LDS 0
#endif

__device__ __forceinline__ int iclampi(int v, int lo, int hi) {
    return v < lo ? lo : (v > hi ? hi : v);
}

// Stage 240*64 f16 (30720 B) of W into LDS, whole block cooperating.
__device__ __forceinline__ void stageW(const _Float16* __restrict__ Wh,
                                       _Float16* __restrict__ ldsW) {
#if USE_ASYNC_LDS
    for (int i = threadIdx.x; i < (240 * 64) / 8; i += 256) {       // 16B chunks
        __builtin_amdgcn_global_load_async_to_lds_b128(
            (__attribute__((address_space(1))) v4i*)(Wh + (size_t)i * 8),
            (__attribute__((address_space(3))) v4i*)(ldsW + (size_t)i * 8),
            0, 0);
    }
    __builtin_amdgcn_s_wait_asynccnt(0);   // this wave's copies done
    __syncthreads();                        // all waves' copies done
#else
    for (int i = threadIdx.x; i < (240 * 64) / 8; i += 256)
        ((float4*)ldsW)[i] = ((const float4*)Wh)[i];
    __syncthreads();
#endif
}

// Block-wide reduction, blockDim.x == 256.  OP: 0=sum 1=min 2=max
template <int OP>
__device__ float bred(float v, float* sh) {
    int t = threadIdx.x;
    sh[t] = v;
    __syncthreads();
    for (int s = 128; s > 0; s >>= 1) {
        if (t < s) {
            float a = sh[t], b = sh[t + s];
            sh[t] = (OP == 0) ? (a + b) : (OP == 1) ? fminf(a, b) : fmaxf(a, b);
        }
        __syncthreads();
    }
    float r = sh[0];
    __syncthreads();
    return r;
}

// ---------------------------------------------------------------- pass 1
// Per (layer,b,c) tile: min/max of border-zeroed data.
__global__ void k_tileMinMax(const float* __restrict__ in, float* __restrict__ stats) {
    int tile = blockIdx.x;                       // l*2048 + b*256 + c
    const float* p = in + (size_t)tile * NPIX;
    float mn = 3.4e38f, mx = -3.4e38f;
    for (int i = threadIdx.x; i < NPIX; i += 256) {
        int r = i / 56, c = i % 56;
        float v = p[i];
        if (r == 0 || r == 55 || c == 0 || c == 55) v = 0.f;
        mn = fminf(mn, v);
        mx = fmaxf(mx, v);
    }
    __shared__ float sh[256];
    mn = bred<1>(mn, sh);
    mx = bred<2>(mx, sh);
    if (threadIdx.x == 0) {
        stats[(size_t)tile * 16 + 0] = mn;
        stats[(size_t)tile * 16 + 1] = mx;
    }
}

// ---------------------------------------------------------------- pass 2
// Per layer: global max of normalized x (global min is exactly 0).
__global__ void k_layerMax(const float* __restrict__ stats, float* __restrict__ mxg) {
    int l = blockIdx.x;
    float m = 0.f;
    for (int t = threadIdx.x; t < 2048; t += 256) {
        size_t tile = (size_t)l * 2048 + t;
        float mn = stats[tile * 16 + 0], mx = stats[tile * 16 + 1];
        float r = mx - mn;
        m = fmaxf(m, 256.f * r / (r + EPSV));
    }
    __shared__ float sh[256];
    m = bred<2>(m, sh);
    if (threadIdx.x == 0) mxg[l] = m;
}

// ---------------------------------------------------------------- pass 3
// Per tile: histogram vs layer-global edges, -log, collapse the whole
// weight chain into scalars {min1, 1/r1, F} + hlog[11].
__global__ void k_tileStats(const float* __restrict__ in, float* __restrict__ stats,
                            const float* __restrict__ mxg) {
    int tile = blockIdx.x;
    int l    = tile / 2048;
    __shared__ float xs_[NPIX];
    __shared__ float red[256];
    __shared__ int   hist[NBINS];
    __shared__ float hlog[NBINS];

    float tmn = stats[(size_t)tile * 16 + 0];
    float tmx = stats[(size_t)tile * 16 + 1];
    float xsc = 256.f / (tmx - tmn + EPSV);
    const float* p = in + (size_t)tile * NPIX;

    if (threadIdx.x < NBINS) hist[threadIdx.x] = 0;
    for (int i = threadIdx.x; i < NPIX; i += 256) {
        int r = i / 56, c = i % 56;
        float v = p[i];
        if (r == 0 || r == 55 || c == 0 || c == 55) v = 0.f;
        xs_[i] = (v - tmn) * xsc;
    }
    __syncthreads();

    float gmax     = mxg[l];
    float binscale = (float)NBINS / (gmax + 1e-20f);   // edges uniform on [0, gmax]
    for (int i = threadIdx.x; i < NPIX; i += 256) {
        int b = iclampi((int)floorf(xs_[i] * binscale), 0, NBINS - 1);
        atomicAdd(&hist[b], 1);
    }
    __syncthreads();
    if (threadIdx.x < NBINS)
        hlog[threadIdx.x] = -logf((float)hist[threadIdx.x] / (float)NPIX + 1e-4f);
    __syncthreads();

    // reductions over dst0 = hlog[hidx]
    float lmn = 3.4e38f, lmx = -3.4e38f, ls = 0.f, lss = 0.f;
    for (int i = threadIdx.x; i < NPIX; i += 256) {
        int h   = iclampi((int)(xs_[i] * (10.f / 256.f)), 0, NBINS - 1);
        float d = hlog[h];
        lmn = fminf(lmn, d); lmx = fmaxf(lmx, d);
        ls += d; lss += d * d;
    }
    float min1   = bred<1>(lmn, red);
    float max1   = bred<2>(lmx, red);
    float sum0   = bred<0>(ls,  red);
    float sumsq0 = bred<0>(lss, red);

    const float Nf = (float)NPIX;
    float inv1   = 1.f / (max1 - min1 + EPSV);
    float m1max  = (max1 - min1) * inv1;
    float m1mean = (sum0 / Nf - min1) * inv1;
    float w1     = (m1max - m1mean); w1 *= w1;
    float sumsq1 = (sumsq0 - 2.f * min1 * sum0 + Nf * min1 * min1) * inv1 * inv1;
    float max2   = w1 * w1 * m1max * m1max;
    float mean2  = w1 * w1 * sumsq1 / Nf;
    float w2     = (max2 - mean2); w2 *= w2;
    float max3   = w2 * max2;
    float K      = w2 * w1 * w1 / (max3 + EPSV);   // dst4 = K * dst1^2

    // interior (2-px margin) reductions of t = dst1^2
    float bmx = 0.f, bs = 0.f;
    for (int i = threadIdx.x; i < NPIX; i += 256) {
        int r = i / 56, c = i % 56;
        if (r >= 2 && r <= 53 && c >= 2 && c <= 53) {
            int h   = iclampi((int)(xs_[i] * (10.f / 256.f)), 0, NBINS - 1);
            float t = (hlog[h] - min1) * inv1;
            t *= t;
            bmx = fmaxf(bmx, t);
            bs += t * t;
        }
    }
    float maxB  = bred<2>(bmx, red);
    float sumB  = bred<0>(bs,  red);
    float max5  = K * K * maxB * maxB;
    float mean5 = K * K * sumB / Nf;
    float w3    = (max5 - mean5); w3 *= w3;
    float F     = w3 * K * K;                       // dst6 = F * dst1^4 (interior)

    if (threadIdx.x == 0) {
        stats[(size_t)tile * 16 + 2] = min1;
        stats[(size_t)tile * 16 + 3] = inv1;
        stats[(size_t)tile * 16 + 4] = F;
    }
    if (threadIdx.x < NBINS)
        stats[(size_t)tile * 16 + 5 + threadIdx.x] = hlog[threadIdx.x];
}

// ---------------------------------------------------------------- pass 4
// Per (layer,b): recompute dst6 per channel from stored scalars, sum over C.
__global__ void k_channelSum(const float* __restrict__ in, const float* __restrict__ stats,
                             float* __restrict__ rmap) {
    int lb = blockIdx.x;                         // 0..79
    float acc[13];
#pragma unroll
    for (int j = 0; j < 13; ++j) acc[j] = 0.f;

    __shared__ float s_hlog[NBINS];
    __shared__ float s_sc[5];

    for (int c = 0; c < 256; ++c) {
        size_t tile = (size_t)lb * 256 + c;
        __syncthreads();
        if (threadIdx.x < NBINS) s_hlog[threadIdx.x] = stats[tile * 16 + 5 + threadIdx.x];
        if (threadIdx.x >= 32 && threadIdx.x < 37)
            s_sc[threadIdx.x - 32] = stats[tile * 16 + (threadIdx.x - 32)];
        __syncthreads();

        float tmn  = s_sc[0];
        float xsc  = 256.f / (s_sc[1] - tmn + EPSV);
        float min1 = s_sc[2], inv1 = s_sc[3], F = s_sc[4];
        const float* p = in + tile * NPIX;
        if (c + 1 < 256)   // hide next tile's HBM latency (global_prefetch_b8)
            __builtin_prefetch(p + NPIX + threadIdx.x * 4, 0, 1);
#pragma unroll
        for (int j = 0; j < 13; ++j) {
            int i = threadIdx.x + 256 * j;
            if (i >= NPIX) break;
            int r = i / 56, cc = i % 56;
            float v = p[i];
            if (r == 0 || r == 55 || cc == 0 || cc == 55) v = 0.f;
            float xv = (v - tmn) * xsc;
            int h    = iclampi((int)(xv * (10.f / 256.f)), 0, NBINS - 1);
            float t  = (s_hlog[h] - min1) * inv1;
            t *= t;
            float val = F * t * t;
            if (r < 2 || r > 53 || cc < 2 || cc > 53) val = 0.f;
            acc[j] += val;
        }
    }
#pragma unroll
    for (int j = 0; j < 13; ++j) {
        int i = threadIdx.x + 256 * j;
        if (i < NPIX) rmap[(size_t)lb * NPIX + i] = acc[j];
    }
}

// ---------------------------------------------------------------- resize setup
// Bilinear interp matrix W[240][64] (f16, K-padded with zeros), half-pixel
// centers with edge clamp (jax.image.resize 'bilinear' for 56 -> 240).
__global__ void k_buildW(_Float16* __restrict__ Wh) {
    int idx = blockIdx.x * 256 + threadIdx.x;
    if (idx >= 240 * 64) return;
    int i = idx / 64, j = idx % 64;
    float w = 0.f;
    if (j < 56) {
        float s  = (i + 0.5f) * (56.0f / 240.0f) - 0.5f;
        float fs = floorf(s);
        int   j0 = (int)fs;
        float f  = s - fs;
        int ja = iclampi(j0, 0, 55), jb = iclampi(j0 + 1, 0, 55);
        if (j == ja) w += 1.0f - f;
        if (j == jb) w += f;
    }
    Wh[idx] = (_Float16)w;
}

// rmap [80][56][56] f32 -> Xh [80][64][64] f16 zero-padded
__global__ void k_convX(const float* __restrict__ rmap, _Float16* __restrict__ Xh) {
    int idx = blockIdx.x * 256 + threadIdx.x;      // 80*4096
    if (idx >= NMAPS * 4096) return;
    int img = idx >> 12, rp = idx & 4095;
    int r = rp >> 6, c = rp & 63;
    float v = 0.f;
    if (r < 56 && c < 56) v = rmap[(size_t)img * NPIX + r * 56 + c];
    Xh[idx] = (_Float16)v;
}

// ---------------------------------------------------------------- WMMA helpers
// ISA 7.12.2 layouts (wave32): A 16x32 f16 row-per-lane with K split across
// lane halves; B 32x16 f16 col-per-lane; C/D 16x16 f32 = 8 VGPRs, M split.
__device__ __forceinline__ v16h loadA_rowmajor(const _Float16* __restrict__ base,
                                               int m, int ld, int k0, int half) {
    v16h a;
#pragma unroll
    for (int v = 0; v < 8; ++v) {
        int kb = (v < 4) ? ((half ? 8 : 0) + 2 * v) : ((half ? 24 : 16) + 2 * (v - 4));
        a[2 * v]     = base[(size_t)m * ld + k0 + kb];
        a[2 * v + 1] = base[(size_t)m * ld + k0 + kb + 1];
    }
    return a;
}

// ---------------------------------------------------------------- GEMM1
// T[img][240][64] = W[240][64(K)] x X[img][64][64]   (K-pad columns are 0)
// W is staged into LDS once per block via async global->LDS copies.
__global__ void k_gemm1(const _Float16* __restrict__ Wh, const _Float16* __restrict__ Xh,
                        _Float16* __restrict__ Th) {
    __shared__ __align__(16) _Float16 ldsW[240 * 64];
    stageW(Wh, ldsW);

    int wave = blockIdx.x * 8 + (threadIdx.x >> 5);
    if (wave >= NMAPS * 15 * 4) return;            // wave-uniform
    int img = wave / 60;
    int rem = wave % 60;
    int mi = rem / 4, nj = rem % 4;
    int lane = threadIdx.x & 31;
    int half = lane >> 4, row16 = lane & 15;
    const _Float16* X = Xh + (size_t)img * 64 * 64;

    v8f c = {};
#pragma unroll
    for (int k0 = 0; k0 < 64; k0 += 32) {
        v16h a = loadA_rowmajor(ldsW, mi * 16 + row16, 64, k0, half);
        v16h b;
        int n = nj * 16 + row16;
#pragma unroll
        for (int v = 0; v < 8; ++v) {
            int kb = (v < 4) ? ((half ? 8 : 0) + 2 * v) : ((half ? 24 : 16) + 2 * (v - 4));
            b[2 * v]     = X[(size_t)(k0 + kb) * 64 + n];
            b[2 * v + 1] = X[(size_t)(k0 + kb + 1) * 64 + n];
        }
        c = __builtin_amdgcn_wmma_f32_16x16x32_f16(false, a, false, b, (short)0, c,
                                                   false, false);
    }
    _Float16* T = Th + (size_t)img * 240 * 64;
#pragma unroll
    for (int i = 0; i < 8; ++i) {
        int m = mi * 16 + (half ? 8 : 0) + i;
        T[(size_t)m * 64 + nj * 16 + row16] = (_Float16)c[i];
    }
}

// ---------------------------------------------------------------- GEMM2
// out[img][240][240] = T[img][240][64] x W^T  (B[k][n] = W[n][k], W in LDS)
__global__ void k_gemm2(const _Float16* __restrict__ Th, const _Float16* __restrict__ Wh,
                        float* __restrict__ out240) {
    __shared__ __align__(16) _Float16 ldsW[240 * 64];
    stageW(Wh, ldsW);

    int wave = blockIdx.x * 8 + (threadIdx.x >> 5);
    if (wave >= NMAPS * 15 * 15) return;           // wave-uniform
    int img = wave / 225;
    int rem = wave % 225;
    int mi = rem / 15, nj = rem % 15;
    int lane = threadIdx.x & 31;
    int half = lane >> 4, row16 = lane & 15;
    const _Float16* T = Th + (size_t)img * 240 * 64;

    v8f c = {};
#pragma unroll
    for (int k0 = 0; k0 < 64; k0 += 32) {
        v16h a = loadA_rowmajor(T, mi * 16 + row16, 64, k0, half);
        v16h b;
        int n = nj * 16 + row16;
#pragma unroll
        for (int v = 0; v < 8; ++v) {
            int kb = (v < 4) ? ((half ? 8 : 0) + 2 * v) : ((half ? 24 : 16) + 2 * (v - 4));
            b[2 * v]     = ldsW[(size_t)n * 64 + k0 + kb];    // W^T gather from LDS
            b[2 * v + 1] = ldsW[(size_t)n * 64 + k0 + kb + 1];
        }
        c = __builtin_amdgcn_wmma_f32_16x16x32_f16(false, a, false, b, (short)0, c,
                                                   false, false);
    }
    float* O = out240 + (size_t)img * RPIX;
#pragma unroll
    for (int i = 0; i < 8; ++i) {
        int m = mi * 16 + (half ? 8 : 0) + i;
        O[(size_t)m * 240 + nj * 16 + row16] = c[i];
    }
}

// ---------------------------------------------------------------- fusion
__global__ void k_mapReduce(const float* __restrict__ out240, float* __restrict__ mapred) {
    int map = blockIdx.x;
    const float* p = out240 + (size_t)map * RPIX;
    float mn = 3.4e38f, mx = -3.4e38f, sm = 0.f;
    for (int i = threadIdx.x; i < RPIX; i += 256) {
        float v = p[i];
        mn = fminf(mn, v); mx = fmaxf(mx, v); sm += v;
    }
    __shared__ float sh[256];
    mn = bred<1>(mn, sh);
    mx = bred<2>(mx, sh);
    sm = bred<0>(sm, sh);
    if (threadIdx.x == 0) {
        mapred[map * 4 + 0] = mn;
        mapred[map * 4 + 1] = mx;
        mapred[map * 4 + 2] = sm;
    }
}

// per map: double-minmax + (max-mean)^2 weight -> affine coefficients
__global__ void k_mapCoef(const float* __restrict__ mapred, float* __restrict__ coef) {
    int map = threadIdx.x;
    if (map >= NMAPS) return;
    float mn = mapred[map * 4 + 0], mx = mapred[map * 4 + 1], sm = mapred[map * 4 + 2];
    float r = mx - mn;
    float invr = 1.f / (r + EPSV);
    float max1 = r * invr;
    float mean1 = (sm / (float)RPIX - mn) * invr;
    float inv2 = 1.f / (max1 + EPSV);
    float max2 = max1 * inv2, mean2 = mean1 * inv2;
    float w = (max2 - mean2); w *= w;
    float A = w * invr * inv2;
    coef[map * 2 + 0] = A;
    coef[map * 2 + 1] = -A * mn;
}

__global__ void k_fuse(const float* __restrict__ out240, const float* __restrict__ coef,
                       float* __restrict__ fused) {
    int idx = blockIdx.x * 256 + threadIdx.x;      // 5 * 57600
    if (idx >= 5 * RPIX) return;
    int g = idx / RPIX, p = idx % RPIX;
    float s = 0.f;
#pragma unroll
    for (int h = 0; h < 2; ++h)
        for (int b = 0; b < 8; ++b) {
            int map = (2 * g + h) * 8 + b;
            s += coef[map * 2 + 0] * out240[(size_t)map * RPIX + p] + coef[map * 2 + 1];
        }
    fused[idx] = s;
}

__global__ void k_groupMinMax(const float* __restrict__ fused, float* __restrict__ gmm) {
    int g = blockIdx.x;
    const float* p = fused + (size_t)g * RPIX;
    float mn = 3.4e38f, mx = -3.4e38f;
    for (int i = threadIdx.x; i < RPIX; i += 256) {
        float v = p[i];
        mn = fminf(mn, v); mx = fmaxf(mx, v);
    }
    __shared__ float sh[256];
    mn = bred<1>(mn, sh);
    mx = bred<2>(mx, sh);
    if (threadIdx.x == 0) { gmm[g * 2 + 0] = mn; gmm[g * 2 + 1] = mx; }
}

// groups (minmax*255) -> d_out[57600..], SAL = sum over groups -> ws
__global__ void k_groupsOutSal(const float* __restrict__ fused, const float* __restrict__ gmm,
                               float* __restrict__ d_out, float* __restrict__ sal) {
    int p = blockIdx.x * 256 + threadIdx.x;
    if (p >= RPIX) return;
    float s = 0.f;
#pragma unroll
    for (int g = 0; g < 5; ++g) {
        float mn = gmm[g * 2 + 0], mx = gmm[g * 2 + 1];
        float v = (fused[(size_t)g * RPIX + p] - mn) * (255.f / (mx - mn + EPSV));
        d_out[RPIX + (size_t)g * RPIX + p] = v;
        s += v;
    }
    sal[p] = s;
}

__global__ void k_salMinMax(const float* __restrict__ sal, float* __restrict__ salmm) {
    float mn = 3.4e38f, mx = -3.4e38f;
    for (int i = threadIdx.x; i < RPIX; i += 256) {
        float v = sal[i];
        mn = fminf(mn, v); mx = fmaxf(mx, v);
    }
    __shared__ float sh[256];
    mn = bred<1>(mn, sh);
    mx = bred<2>(mx, sh);
    if (threadIdx.x == 0) { salmm[0] = mn; salmm[1] = mx; }
}

// s = minmax(exp(minmax(SAL)))
__global__ void k_final(const float* __restrict__ sal, const float* __restrict__ salmm,
                        float* __restrict__ d_out) {
    int p = blockIdx.x * 256 + threadIdx.x;
    if (p >= RPIX) return;
    float mn = salmm[0], mx = salmm[1];
    float inv = 1.f / (mx - mn + EPSV);
    float u = (sal[p] - mn) * inv;                 // in [0, umax], min exactly 0
    float umax = (mx - mn) * inv;
    float e = expf(u);
    d_out[p] = (e - 1.f) / (expf(umax) - 1.f + EPSV);
}

// ================================================================ launch
extern "C" void kernel_launch(void* const* d_in, const int* in_sizes, int n_in,
                              void* d_out, int out_size, void* d_ws, size_t ws_size,
                              hipStream_t stream) {
    const float* in = (const float*)d_in[0];       // [10,8,256,56,56] f32
    float* out = (float*)d_out;                    // s(57600) then groups(288000)

    char* base = (char*)d_ws;
    size_t off = 0;
    auto alloc = [&](size_t bytes) -> char* {
        char* p = base + off;
        off = (off + bytes + 255) & ~(size_t)255;
        return p;
    };

    float*    stats  = (float*)alloc((size_t)20480 * 16 * 4);
    float*    mxg    = (float*)alloc(10 * 4);
    float*    rmap   = (float*)alloc((size_t)NMAPS * NPIX * 4);
    float*    out240 = (float*)alloc((size_t)NMAPS * RPIX * 4);
    float*    mapred = (float*)alloc(NMAPS * 4 * 4);
    float*    coef   = (float*)alloc(NMAPS * 2 * 4);
    float*    fused  = (float*)alloc((size_t)5 * RPIX * 4);
    float*    gmm    = (float*)alloc(5 * 2 * 4);
    float*    sal    = (float*)alloc((size_t)RPIX * 4);
    float*    salmm  = (float*)alloc(2 * 4);
    _Float16* Wh     = (_Float16*)alloc((size_t)240 * 64 * 2);
    _Float16* Xh     = (_Float16*)alloc((size_t)NMAPS * 64 * 64 * 2);
    _Float16* Th     = (_Float16*)alloc((size_t)NMAPS * 240 * 64 * 2);
    if (off > ws_size) return;                     // scratch too small: no-op

    // rarity pipeline (3 passes over the 257 MB input; bandwidth bound)
    k_tileMinMax<<<20480, 256, 0, stream>>>(in, stats);
    k_layerMax  <<<10,    256, 0, stream>>>(stats, mxg);
    k_tileStats <<<20480, 256, 0, stream>>>(in, stats, mxg);
    k_channelSum<<<NMAPS, 256, 0, stream>>>(in, stats, rmap);

    // bilinear resize 56->240 as two f16 WMMA GEMMs per map
    k_buildW<<<(240 * 64 + 255) / 256, 256, 0, stream>>>(Wh);
    k_convX <<<(NMAPS * 4096 + 255) / 256, 256, 0, stream>>>(rmap, Xh);
    k_gemm1 <<<(NMAPS * 15 * 4 + 7) / 8,  256, 0, stream>>>(Wh, Xh, Th);
    k_gemm2 <<<(NMAPS * 15 * 15 + 7) / 8, 256, 0, stream>>>(Th, Wh, out240);

    // Itti fusion + final normalization
    k_mapReduce  <<<NMAPS, 256, 0, stream>>>(out240, mapred);
    k_mapCoef    <<<1, 128, 0, stream>>>(mapred, coef);
    k_fuse       <<<(5 * RPIX + 255) / 256, 256, 0, stream>>>(out240, coef, fused);
    k_groupMinMax<<<5, 256, 0, stream>>>(fused, gmm);
    k_groupsOutSal<<<RPIX / 256, 256, 0, stream>>>(fused, gmm, out, sal);
    k_salMinMax  <<<1, 256, 0, stream>>>(sal, salmm);
    k_final      <<<RPIX / 256, 256, 0, stream>>>(sal, salmm, out);
}
